// Decoder_81097572483397
// MI455X (gfx1250) — compile-verified, matrix operand order checked by reading, not run on hardware
//
#include <hip/hip_runtime.h>
#include <hip/hip_bf16.h>
#include <math.h>
#include <stdint.h>

// ---------------------------------------------------------------------------
// Decoder scan for MI455X (gfx1250, wave32).
//   B=256, T-1=127, E=128, D=128.
// Kernel 1: enc_proj = X_encoded @ W1_e + b1  (WMMA bf16; persistent blocks,
//           weight fragments in registers, ASYNC global->LDS double-buffered
//           tile staging overlapped with WMMA compute)
// Kernel 2: per-WG scan, 16 batch rows / WG. Weight B-fragments and per-step
//           A-fragments held in registers; hardware v_tanh_f32 transcendentals.
// ---------------------------------------------------------------------------

typedef __attribute__((ext_vector_type(16))) __bf16 v16bf;
typedef __attribute__((ext_vector_type(8)))  float  v8f;

#define B_TOT   256
#define TM1     127
#define E_DIM   128
#define D_DIM   128
#define G_DIM   512            // 4*D
#define ROWS_PER_WG 16
#define K2_THREADS 512         // 16 waves of 32

// ----------------------------- fast transcendentals ------------------------
__device__ __forceinline__ float tanh_fast(float x) {
#if __has_builtin(__builtin_amdgcn_tanhf)
  return __builtin_amdgcn_tanhf(x);
#else
  float r;
  // gfx1250 TRANS32 op; TRANS RAW needs 1 slot before use (ISA 7.4).
  asm("v_tanh_f32 %0, %1\n\ts_delay_alu instid0(TRANS32_DEP_1)"
      : "=v"(r) : "v"(x));
  return r;
#endif
}
__device__ __forceinline__ float sigmoid_fast(float x) {
  return 0.5f * tanh_fast(0.5f * x) + 0.5f;   // one TRANS op instead of two
}

// ----------------------- async global->LDS primitives ----------------------
// Each lane copies 16 bytes: LDS[lds_off] = MEM[gaddr]. Tracked by ASYNCcnt.
__device__ __forceinline__ void async_ld_b128(uint32_t lds_off, const void* g) {
  asm volatile("global_load_async_to_lds_b128 %0, %1, off"
               :: "v"(lds_off), "v"((uint64_t)(uintptr_t)g) : "memory");
}
__device__ __forceinline__ void wait_async0() {
#if __has_builtin(__builtin_amdgcn_s_wait_asynccnt)
  __builtin_amdgcn_s_wait_asynccnt(0);
#else
  asm volatile("s_wait_asynccnt 0x0" ::: "memory");
#endif
}

// ----------------------------- wave reductions -----------------------------
__device__ __forceinline__ float wsum(float v) {
#pragma unroll
  for (int s = 16; s > 0; s >>= 1) v += __shfl_xor(v, s, 32);
  return v;
}
__device__ __forceinline__ float wmaxr(float v) {
#pragma unroll
  for (int s = 16; s > 0; s >>= 1) v = fmaxf(v, __shfl_xor(v, s, 32));
  return v;
}

// ------------------------ WMMA fragment constructors ------------------------
// 16-bit A (16x32, MxK) lane layout (ISA 7.12.2): lanes 0-15 hold M=lane,
// K groups {0..7,16..23}; lanes 16-31 hold M=lane-16, K groups {8..15,24..31}.
// v16bf element pair (2i,2i+1) = one VGPR (low,high half).
__device__ __forceinline__ v16bf a_frag_f32(const float* __restrict__ src,
                                            int ldk, int lane, int k0) {
  const int m  = lane & 15;
  const int kb = k0 + ((lane >> 4) ? 8 : 0);
  const float* r = src + m * ldk;
  v16bf a;
#pragma unroll
  for (int i = 0; i < 4; ++i) {
    a[2*i]     = (__bf16)r[kb + 2*i];
    a[2*i + 1] = (__bf16)r[kb + 2*i + 1];
  }
#pragma unroll
  for (int i = 0; i < 4; ++i) {
    a[8 + 2*i] = (__bf16)r[kb + 16 + 2*i];
    a[9 + 2*i] = (__bf16)r[kb + 16 + 2*i + 1];
  }
  return a;
}
// 16-bit B (32x16, KxN): mirror of A with N on the lane axis.
// LDS bf16 source, W stored [k][n]:
__device__ __forceinline__ v16bf b_frag(const __bf16* __restrict__ W,
                                        int ldn, int lane, int k0, int n0) {
  const int n  = n0 + (lane & 15);
  const int kb = k0 + ((lane >> 4) ? 8 : 0);
  v16bf b;
#pragma unroll
  for (int i = 0; i < 4; ++i) {
    b[2*i]     = W[(kb + 2*i)     * ldn + n];
    b[2*i + 1] = W[(kb + 2*i + 1) * ldn + n];
  }
#pragma unroll
  for (int i = 0; i < 4; ++i) {
    b[8 + 2*i] = W[(kb + 16 + 2*i) * ldn + n];
    b[9 + 2*i] = W[(kb + 17 + 2*i) * ldn + n];
  }
  return b;
}
// Global f32 source, W stored [k][n] row-major (one-time register fill):
__device__ __forceinline__ v16bf b_frag_g(const float* __restrict__ W,
                                          int ldn, int lane, int k0, int n0) {
  const int n  = n0 + (lane & 15);
  const int kb = k0 + ((lane >> 4) ? 8 : 0);
  v16bf b;
#pragma unroll
  for (int i = 0; i < 4; ++i) {
    b[2*i]     = (__bf16)W[(kb + 2*i)     * ldn + n];
    b[2*i + 1] = (__bf16)W[(kb + 2*i + 1) * ldn + n];
  }
#pragma unroll
  for (int i = 0; i < 4; ++i) {
    b[8 + 2*i] = (__bf16)W[(kb + 16 + 2*i) * ldn + n];
    b[9 + 2*i] = (__bf16)W[(kb + 17 + 2*i) * ldn + n];
  }
  return b;
}
// Global f32 source, logical B[k][n] = W[n*ldk + k] (transposed use, for W_hh):
__device__ __forceinline__ v16bf b_frag_gT(const float* __restrict__ W,
                                           int ldk, int lane, int k0, int n0) {
  const int n  = n0 + (lane & 15);
  const int kb = k0 + ((lane >> 4) ? 8 : 0);
  const float* r = W + n * ldk;
  v16bf b;
#pragma unroll
  for (int i = 0; i < 4; ++i) {
    b[2*i]     = (__bf16)r[kb + 2*i];
    b[2*i + 1] = (__bf16)r[kb + 2*i + 1];
  }
#pragma unroll
  for (int i = 0; i < 4; ++i) {
    b[8 + 2*i] = (__bf16)r[kb + 16 + 2*i];
    b[9 + 2*i] = (__bf16)r[kb + 16 + 2*i + 1];
  }
  return b;
}

// ---------------------------------------------------------------------------
// Kernel 1: enc_proj[row][n] = sum_k X[row][k] * W1_e[k][n] + b1[n]
//   rows = B*TM1 = 32512 = 127 blocks * 16 tiles * 16 rows.
//   W1_e staged once; B-fragments registered once; X tiles (8 KB f32)
//   double-buffered via GLOBAL_LOAD_ASYNC_TO_LDS_B128 (2 instrs/wave/tile),
//   overlapping the next tile's DMA with the current tile's WMMAs.
// ---------------------------------------------------------------------------
#define K1_TILES 16

__launch_bounds__(256)
__global__ void encproj_kernel(const float* __restrict__ X,
                               const float* __restrict__ W1,
                               const float* __restrict__ b1,
                               float* __restrict__ encp) {
  __shared__ __bf16 sW1e[E_DIM * E_DIM];          // [k][n]  32 KB
  __shared__ float  sXf[2][ROWS_PER_WG * E_DIM];  // [buf][m][k]  2 x 8 KB

  const int tid = threadIdx.x;
  const char* xbytes = (const char*)X;

  // Kick off tile 0 DMA before anything else (overlaps W1_e staging).
  {
    const size_t rb0 = (size_t)blockIdx.x * K1_TILES * ROWS_PER_WG;
    const uint32_t l0 = (uint32_t)(uintptr_t)(&sXf[0][0]) + tid * 16;
    const char* g0 = xbytes + rb0 * E_DIM * 4 + tid * 16;
    async_ld_b128(l0,        g0);
    async_ld_b128(l0 + 4096, g0 + 4096);
  }

  // W1_e = W1 rows [256,384): contiguous at W1 + 256*128
  for (int i = tid; i < E_DIM * E_DIM; i += 256)
    sW1e[i] = (__bf16)W1[2 * D_DIM * E_DIM + i];
  __syncthreads();

  const int wave = tid >> 5;
  const int lane = tid & 31;
  const int n0 = wave * 16;
  const int n  = n0 + (lane & 15);
  const int mhi = (lane >> 4) ? 8 : 0;
  const float bn = b1[n];

  // loop-invariant weight fragments
  v16bf bW[4];
#pragma unroll
  for (int kc = 0; kc < 4; ++kc)
    bW[kc] = b_frag(sW1e, E_DIM, lane, kc * 32, n0);

  for (int it = 0; it < K1_TILES; ++it) {
    // Wait for this tile's DMA (issued last iteration / prologue)...
    wait_async0();
    __syncthreads();   // ...and make sure every wave's slice has landed,
                       // and that buf (it+1)&1 is no longer being read.
    // Launch next tile's DMA into the other buffer; overlaps compute below.
    if (it + 1 < K1_TILES) {
      const size_t rbn = (size_t)(blockIdx.x * K1_TILES + it + 1) * ROWS_PER_WG;
      const uint32_t ln = (uint32_t)(uintptr_t)(&sXf[(it + 1) & 1][0]) + tid * 16;
      const char* gn = xbytes + rbn * E_DIM * 4 + tid * 16;
      async_ld_b128(ln,        gn);
      async_ld_b128(ln + 4096, gn + 4096);
    }

    const float* xt = &sXf[it & 1][0];
    const int rowbase = (blockIdx.x * K1_TILES + it) * ROWS_PER_WG;
    v8f acc = {};
#pragma unroll
    for (int kc = 0; kc < 4; ++kc) {
      v16bf a = a_frag_f32(xt, E_DIM, lane, kc * 32);
      acc = __builtin_amdgcn_wmma_f32_16x16x32_bf16(false, a, false, bW[kc],
                                                    (short)0, acc, false, false);
    }
#pragma unroll
    for (int i = 0; i < 8; ++i) {
      int row = rowbase + i + mhi;
      encp[(size_t)row * E_DIM + n] = acc[i] + bn;
    }
  }
}

// ---------------------------------------------------------------------------
// Kernel 2: the 127-step scan. grid = 16 blocks (16 batch rows each),
// 512 threads = 16 waves. Weight fragments live in registers for all steps.
// LDS: sd,sc,su,su2,sctx,sbeta (6*2048) + sgates (8192) + syt (16).
// ---------------------------------------------------------------------------
#define K2_SMEM_BYTES (20496 * 4)   // 81,984 B

__launch_bounds__(K2_THREADS)
__global__ void scan_kernel(const float* __restrict__ X,
                            const float* __restrict__ yprev,
                            const float* __restrict__ W1,
                            const float* __restrict__ W2,
                            const float* __restrict__ Wih,
                            const float* __restrict__ Whh,
                            const float* __restrict__ bih,
                            const float* __restrict__ bhh,
                            const float* __restrict__ Wf,
                            const float* __restrict__ bfv,
                            const float* __restrict__ Wff,
                            const float* __restrict__ bff,
                            const float* __restrict__ encp,
                            float* __restrict__ out) {
  extern __shared__ char smem[];
  float* sd     = (float*)smem;              // [16][128]
  float* sc     = sd     + 16 * 128;         // [16][128]
  float* su     = sc     + 16 * 128;         // [16][128]  d@W1_d partial
  float* su2    = su     + 16 * 128;         // [16][128]  c@W1_c partial
  float* sctx   = su2    + 16 * 128;         // [16][128]
  float* sbeta  = sctx   + 16 * 128;         // [16][128] scores, then beta
  float* sgates = sbeta  + 16 * 128;         // [16][512]
  float* syt    = sgates + 16 * 512;         // [16]

  const int tid  = threadIdx.x;
  const int wave = tid >> 5;
  const int lane = tid & 31;
  const int b0   = blockIdx.x * ROWS_PER_WG;
  const int m    = wave;                     // per-wave batch row for B/C/D
  const bool loA = (wave < 8);               // d-half vs c-half of Phase A

  // ---- one-time: loop-invariant weight fragments -> registers ----
  // Phase A: u = d@W1_d + c@W1_c.  Waves 0-7: W1_d (rows 0..127), n0=16*wave.
  //                                Waves 8-15: W1_c (rows 128..255).
  const int nA = (loA ? wave : wave - 8) * 16;
  const float* WA = loA ? W1 : (W1 + D_DIM * E_DIM);
  v16bf bA[4];
#pragma unroll
  for (int kc = 0; kc < 4; ++kc)
    bA[kc] = b_frag_g(WA, E_DIM, lane, kc * 32, nA);

  // Phase E: gates = d @ W_hh^T ; logical B[k][j] = W_hh[j][k].
  // 32 n-tiles over G=512, 2 per wave, K=128 -> 4 fragments per tile.
  v16bf bE[2][4];
  float wjr[2], bjr[2];
#pragma unroll
  for (int p = 0; p < 2; ++p) {
    const int n0 = (wave * 2 + p) * 16;
#pragma unroll
    for (int kc = 0; kc < 4; ++kc)
      bE[p][kc] = b_frag_gT(Whh, D_DIM, lane, kc * 32, n0);
    const int j = n0 + (lane & 15);
    wjr[p] = Wih[j];
    bjr[p] = bih[j] + bhh[j];
  }
  // Per-lane slices of W2 / Wf (e = lane + 32q), plus scalars.
  float w2q[4], wfq[4];
#pragma unroll
  for (int q = 0; q < 4; ++q) {
    w2q[q] = W2[lane + 32 * q];
    wfq[q] = Wf[lane + 32 * q];
  }
  const float wf_y = Wf[E_DIM];
  const float bf0  = bfv[0];

  // ---- LDS init ----
  for (int i = tid; i < 16 * 128; i += K2_THREADS) {
    sd[i] = 0.0f; sc[i] = 0.0f; sctx[i] = 0.0f;
  }
  __syncthreads();

  const size_t xbase = (size_t)(b0 + m) * TM1 * E_DIM;   // this wave's row

  for (int t = 0; t < TM1; ++t) {
    // ---- per-step A-fragments of d (all waves; reused in A and E) and of
    //      c (waves 8-15). sd/sc are stable until Phase F. ----
    v16bf aD[4], aS[4];
#pragma unroll
    for (int kc = 0; kc < 4; ++kc) {
      aD[kc] = a_frag_f32(sd, 128, lane, kc * 32);
      aS[kc] = loA ? aD[kc] : a_frag_f32(sc, 128, lane, kc * 32);
    }

    // ---- Phase A: partial u on all 16 waves ----
    {
      v8f acc = {};
#pragma unroll
      for (int kc = 0; kc < 4; ++kc)
        acc = __builtin_amdgcn_wmma_f32_16x16x32_bf16(false, aS[kc], false, bA[kc],
                                                      (short)0, acc, false, false);
      float* dstu = loA ? su : su2;
      const int n = nA + (lane & 15);
      const int mhi = (lane >> 4) ? 8 : 0;
#pragma unroll
      for (int i = 0; i < 8; ++i) dstu[(i + mhi) * 128 + n] = acc[i];
    }
    __syncthreads();

    // ---- Fused B/C/D (per-wave, m = wave): scores -> softmax -> context ----
    // B: scores[tt] = sum_e W2[e]*tanh(encp[m,tt,e] + u[m,e])   (dominant)
    float uq[4];
#pragma unroll
    for (int q = 0; q < 4; ++q)
      uq[q] = su[m * 128 + lane + 32 * q] + su2[m * 128 + lane + 32 * q];
    for (int tt = 0; tt < TM1; ++tt) {
      const float* ep = encp + xbase + (size_t)tt * E_DIM;
      float acc = 0.0f;
#pragma unroll
      for (int q = 0; q < 4; ++q)
        acc += w2q[q] * tanh_fast(ep[lane + 32 * q] + uq[q]);
      acc = wsum(acc);
      if (lane == 0) sbeta[m * 128 + tt] = acc;  // (+b2 dropped: softmax-invariant)
    }
    // C: softmax over tt (same wave wrote scores; DS is in-order per wave)
    {
      float v[4], ex[4];
      float mx = -3.4e38f;
#pragma unroll
      for (int q = 0; q < 4; ++q) {
        const int tt = lane + 32 * q;
        v[q] = (tt < TM1) ? sbeta[m * 128 + tt] : -3.4e38f;
        mx = fmaxf(mx, v[q]);
      }
      mx = wmaxr(mx);
      float s = 0.0f;
#pragma unroll
      for (int q = 0; q < 4; ++q) {
        const int tt = lane + 32 * q;
        ex[q] = (tt < TM1) ? __expf(v[q] - mx) : 0.0f;
        s += ex[q];
      }
      s = wsum(s);
      const float inv = 1.0f / s;
#pragma unroll
      for (int q = 0; q < 4; ++q) {
        const int tt = lane + 32 * q;
        if (tt < TM1) sbeta[m * 128 + tt] = ex[q] * inv;
      }
    }
    // D: context[m][e] = sum_tt beta*X ; then y_tilde[m]
    {
      float accc[4] = {0.0f, 0.0f, 0.0f, 0.0f};
      for (int tt = 0; tt < TM1; ++tt) {
        const float bv = sbeta[m * 128 + tt];     // same-wave DS broadcast
        const float* xp = X + xbase + (size_t)tt * E_DIM;
#pragma unroll
        for (int ec = 0; ec < 4; ++ec)
          accc[ec] += bv * xp[ec * 32 + lane];
      }
      float a2 = 0.0f;
#pragma unroll
      for (int ec = 0; ec < 4; ++ec) {
        sctx[m * 128 + ec * 32 + lane] = accc[ec];
        a2 += accc[ec] * wfq[ec];
      }
      a2 = wsum(a2);
      if (lane == 0)
        syt[m] = a2 + yprev[(size_t)(b0 + m) * TM1 + t] * wf_y + bf0;
    }
    __syncthreads();

    // ---- Phase E: gates = d @ W_hh^T + y_tilde*W_ih + (b_ih+b_hh) ----
#pragma unroll
    for (int p = 0; p < 2; ++p) {
      v8f acc = {};
#pragma unroll
      for (int kc = 0; kc < 4; ++kc)
        acc = __builtin_amdgcn_wmma_f32_16x16x32_bf16(false, aD[kc], false, bE[p][kc],
                                                      (short)0, acc, false, false);
      const int j = (wave * 2 + p) * 16 + (lane & 15);
      const int mhi = (lane >> 4) ? 8 : 0;
#pragma unroll
      for (int i = 0; i < 8; ++i) {
        const int mm = i + mhi;
        sgates[mm * G_DIM + j] = acc[i] + syt[mm] * wjr[p] + bjr[p];
      }
    }
    __syncthreads();

    // ---- Phase F: LSTM update ----
#pragma unroll
    for (int p = 0; p < 4; ++p) {
      const int idx = tid + p * K2_THREADS;     // 2048 = 16*128
      const int mm = idx >> 7, k = idx & 127;
      const float gi = sgates[mm * G_DIM + k];
      const float gf = sgates[mm * G_DIM + 128 + k];
      const float gg = sgates[mm * G_DIM + 256 + k];
      const float go = sgates[mm * G_DIM + 384 + k];
      const float cn = sigmoid_fast(gf) * sc[idx] + sigmoid_fast(gi) * tanh_fast(gg);
      sc[idx] = cn;
      sd[idx] = sigmoid_fast(go) * tanh_fast(cn);
    }
    __syncthreads();
  }

  // ---- epilogue: out[b] = [d, context] @ Wff + bff  (N_NEXT = 1) ----
  {
    float acc = 0.0f;
#pragma unroll
    for (int q = 0; q < 4; ++q) {
      const int k = lane + 32 * q;
      acc += sd[m * 128 + k] * Wff[k] + sctx[m * 128 + k] * Wff[128 + k];
    }
    acc = wsum(acc);
    if (lane == 0) out[b0 + m] = acc + bff[0];
  }
}

// ---------------------------------------------------------------------------
extern "C" void kernel_launch(void* const* d_in, const int* in_sizes, int n_in,
                              void* d_out, int out_size, void* d_ws, size_t ws_size,
                              hipStream_t stream) {
  const float* X    = (const float*)d_in[0];   // (256,127,128)
  const float* yp   = (const float*)d_in[1];   // (256,127)
  const float* W1   = (const float*)d_in[2];   // (384,128)
  const float* b1   = (const float*)d_in[3];   // (128,)
  const float* W2   = (const float*)d_in[4];   // (128,1)
  // d_in[5] = b2 : softmax-invariant, unused
  const float* Wih  = (const float*)d_in[6];   // (512,1)
  const float* Whh  = (const float*)d_in[7];   // (512,128)
  const float* bih  = (const float*)d_in[8];   // (512,)
  const float* bhh  = (const float*)d_in[9];   // (512,)
  const float* Wf   = (const float*)d_in[10];  // (129,1)
  const float* bfv  = (const float*)d_in[11];  // (1,)
  const float* Wff  = (const float*)d_in[12];  // (256,1)
  const float* bff  = (const float*)d_in[13];  // (1,)
  float* out  = (float*)d_out;                 // (256,1)
  float* encp = (float*)d_ws;                  // 256*127*128 f32 = 16.6 MB

  (void)in_sizes; (void)n_in; (void)out_size; (void)ws_size;

  // Kernel 1: enc_proj GEMM (127 blocks x 16 tiles x 16 rows = 32512 rows).
  encproj_kernel<<<(B_TOT * TM1) / (ROWS_PER_WG * K1_TILES), 256, 0, stream>>>(
      X, W1, b1, encp);

  // Kernel 2: the scan (81,984 B dynamic LDS).
  hipFuncSetAttribute((const void*)scan_kernel,
                      hipFuncAttributeMaxDynamicSharedMemorySize, K2_SMEM_BYTES);
  scan_kernel<<<B_TOT / ROWS_PER_WG, K2_THREADS, K2_SMEM_BYTES, stream>>>(
      X, yp, W1, W2, Wih, Whh, bih, bhh, Wf, bfv, Wff, bff, encp, out);
}
